// MANNBaseEncoder_57578331570951
// MI455X (gfx1250) — matrix operand order repaired and checked by reading, not compiled, not run on hardware
//
#include <hip/hip_runtime.h>
#include <hip/hip_bf16.h>
#include <stdint.h>

// ---------------- problem constants ----------------
#define TT    512
#define BB    64
#define IDIM  256
#define CDIM  512
#define NN    128
#define MM    64
#define KDIM  (IDIM + MM + CDIM)     // 832, = 26 k-blocks of 32
#define GDIM  (4 * CDIM)             // 2048 gate rows
#define ODIM  (CDIM + MM)            // 576 output cols
#define NWG   16
#define NTHR  256
#define EPS   1e-8f

// ---------------- vector types ----------------
typedef __attribute__((ext_vector_type(4)))  float          v4f;
typedef __attribute__((ext_vector_type(8)))  float          v8f;
typedef __attribute__((ext_vector_type(16))) __bf16         v16bf;
typedef __attribute__((ext_vector_type(8)))  unsigned short v8us;
typedef __attribute__((ext_vector_type(16))) unsigned short v16us;

// ---------------- workspace layout (bytes) ----------------
#define OFF_WCAT 0UL                                   // [2048 x 832] bf16
#define SZ_WCAT  ((size_t)GDIM * KDIM * 2)
#define OFF_XH   (OFF_WCAT + SZ_WCAT)                  // [64 x 832] bf16 (x|r|h)
#define SZ_XH    ((size_t)BB * KDIM * 2)
#define OFF_H    (OFF_XH + SZ_XH)                      // [64 x 512] f32
#define SZ_H     ((size_t)BB * CDIM * 4)
#define OFF_C    (OFF_H + SZ_H)                        // [64 x 512] f32
#define SZ_C     SZ_H
#define OFF_MEM  (OFF_C + SZ_C)                        // [64 x 128 x 64] f32
#define SZ_MEM   ((size_t)BB * NN * MM * 4)
#define OFF_CNT  (OFF_MEM + SZ_MEM)                    // barrier counter (64B slot)
#define OFF_BS   (OFF_CNT + 64)                        // [2048] f32 = b_ih + b_hh

// ---------------- helpers ----------------
__device__ __host__ inline unsigned short f2bf(float f) {
    unsigned u = __builtin_bit_cast(unsigned, f);
    unsigned r = u + 0x7FFFu + ((u >> 16) & 1u);       // round-to-nearest-even
    return (unsigned short)(r >> 16);
}

__device__ inline float sigm(float x)  { return 1.f / (1.f + __expf(-x)); }
__device__ inline float splus(float x) { return (x > 20.f) ? x : log1pf(__expf(x)); }
__device__ inline float dot4(v4f a, v4f b) { return a[0]*b[0] + a[1]*b[1] + a[2]*b[2] + a[3]*b[3]; }

// A fragment (16x32 bf16, row-major source): lanes 0-15 hold K {0..7,16..23},
// lanes 16-31 hold K {8..15,24..31} of row M = lane&15  (ISA 7.12.2)
__device__ inline v16bf load_fragA(const unsigned short* rowp, int kb, int hi) {
    const unsigned short* p = rowp + kb * 32 + hi * 8;
    v8us lo = *(const v8us*)(p);
    v8us hh = *(const v8us*)(p + 16);
    v16us full = __builtin_shufflevector(lo, hh, 0,1,2,3,4,5,6,7,8,9,10,11,12,13,14,15);
    return __builtin_bit_cast(v16bf, full);
}
// B fragment (32x16 bf16): B = W^T so lane holds one row of W (col of B);
// lanes 0-15: K {0..7,16..23}, lanes 16-31: K {8..15,24..31} (matches A).
__device__ inline v16bf load_fragB(const unsigned short* rowp, int kb, int hi) {
    const unsigned short* p = rowp + kb * 32 + hi * 8;
    v8us lo = *(const v8us*)(p);
    v8us hh = *(const v8us*)(p + 16);
    v16us full = __builtin_shufflevector(lo, hh, 0,1,2,3,4,5,6,7,8,9,10,11,12,13,14,15);
    return __builtin_bit_cast(v16bf, full);
}

__device__ inline float block_sum(float v, float* red, int tid) {
    red[tid] = v; __syncthreads();
    for (int s = 128; s > 0; s >>= 1) { if (tid < s) red[tid] += red[tid + s]; __syncthreads(); }
    float r = red[0]; __syncthreads();
    return r;
}

__device__ inline void softmax128(float logit, float* dst, float* red, int tid) {
    red[tid] = logit; __syncthreads();
    for (int s = 128; s > 0; s >>= 1) { if (tid < s) red[tid] = fmaxf(red[tid], red[tid + s]); __syncthreads(); }
    float mx = red[0]; __syncthreads();
    float ex = (tid < NN) ? __expf(logit - mx) : 0.f;
    red[tid] = ex; __syncthreads();
    for (int s = 128; s > 0; s >>= 1) { if (tid < s) red[tid] += red[tid + s]; __syncthreads(); }
    float sm = red[0]; __syncthreads();
    if (tid < NN) dst[tid] = ex / sm;
    __syncthreads();
}

// device-scope barrier: monotonically increasing target, counter zeroed by init kernel
__device__ inline void grid_barrier(unsigned* cnt, unsigned want) {
    __syncthreads();
    if (threadIdx.x == 0) {
        __threadfence();
        atomicAdd(cnt, 1u);
        while (__hip_atomic_load(cnt, __ATOMIC_ACQUIRE, __HIP_MEMORY_SCOPE_AGENT) < want) {
            __builtin_amdgcn_s_sleep(1);
        }
    }
    __syncthreads();
}

// ---------------- init kernel: pack weights (bf16), seed state ----------------
__global__ void mann_init_kernel(const float* __restrict__ embs,
                                 const float* __restrict__ W_ih,
                                 const float* __restrict__ W_hh,
                                 const float* __restrict__ b_ih,
                                 const float* __restrict__ b_hh,
                                 const float* __restrict__ h0,
                                 const float* __restrict__ c0,
                                 const float* __restrict__ r0,
                                 const float* __restrict__ mem0,
                                 unsigned short* __restrict__ Wcat,
                                 unsigned short* __restrict__ xh,
                                 float* __restrict__ hG,
                                 float* __restrict__ cG,
                                 float* __restrict__ memG,
                                 float* __restrict__ bsum,
                                 unsigned* __restrict__ cnt) {
    long i = (long)blockIdx.x * blockDim.x + threadIdx.x;
    long stride = (long)gridDim.x * blockDim.x;
    if (i == 0) *cnt = 0u;
    // concatenated weights: row r, col k<320 -> W_ih, else W_hh
    for (long idx = i; idx < (long)GDIM * KDIM; idx += stride) {
        int r = (int)(idx / KDIM), k = (int)(idx % KDIM);
        float v = (k < IDIM + MM) ? W_ih[(long)r * (IDIM + MM) + k]
                                  : W_hh[(long)r * CDIM + (k - IDIM - MM)];
        Wcat[idx] = f2bf(v);
    }
    // x|r|h row for t=0
    for (long idx = i; idx < (long)BB * KDIM; idx += stride) {
        int b = (int)(idx / KDIM), k = (int)(idx % KDIM);
        float v = (k < IDIM) ? embs[(long)b * IDIM + k]
                 : (k < IDIM + MM) ? r0[k - IDIM] : h0[k - IDIM - MM];
        xh[idx] = f2bf(v);
    }
    for (long idx = i; idx < (long)BB * CDIM; idx += stride) {
        hG[idx] = h0[idx & (CDIM - 1)];
        cG[idx] = c0[idx & (CDIM - 1)];
    }
    for (long idx = i; idx < (long)BB * NN * MM; idx += stride)
        memG[idx] = mem0[idx % (NN * MM)];
    for (long idx = i; idx < GDIM; idx += stride)
        bsum[idx] = b_ih[idx] + b_hh[idx];
}

// ---------------- persistent recurrence kernel ----------------
__global__ void __launch_bounds__(NTHR, 1)
mann_step_kernel(const float* __restrict__ embs,
                 const float* __restrict__ bsum,
                 const float* __restrict__ W_kr, const float* __restrict__ b_kr,
                 const float* __restrict__ w_br,
                 const float* __restrict__ W_kw, const float* __restrict__ b_kw,
                 const float* __restrict__ w_bw,
                 const float* __restrict__ W_e,  const float* __restrict__ b_e,
                 const float* __restrict__ W_a,  const float* __restrict__ b_a,
                 const unsigned short* __restrict__ Wcat,
                 unsigned short* __restrict__ xh,
                 float* __restrict__ hG, float* __restrict__ cG,
                 float* __restrict__ memG,
                 unsigned* __restrict__ cnt,
                 float* __restrict__ out) {
    const int tid  = threadIdx.x;
    const int wg   = blockIdx.x;          // 0..15, owns cdim channels [wg*32, wg*32+32)
    const int wave = tid >> 5;            // 0..7 (wave32)
    const int lane = tid & 31;

    __shared__ __align__(16) float gbuf[BB][128];   // gate slice [batch][4g x 32ch] 32KB
    __shared__ __align__(16) float co_s[CDIM];
    __shared__ __align__(16) float kv[MM], ev[MM], av[MM], krv[MM];
    __shared__ __align__(16) float wwv[NN];
    __shared__ __align__(16) float red[NTHR];
    __shared__ float scal[2];

    unsigned bar = 0;

    for (int t = 0; t < TT; ++t) {
        // ========== Phase 1: gates = [x|r|h] @ [W_ih|W_hh]^T (WMMA bf16, f32 acc) ==========
        {
            v8f acc[4] = {};
            const int halfl = lane & 15, hi = lane >> 4;
            const int nt   = wave;                       // N-tile within WG slice
            const int gate = nt >> 1;
            const int rowB = gate * CDIM + wg * 32 + (nt & 1) * 16 + halfl;
            const unsigned short* pB = Wcat + (long)rowB * KDIM;
            for (int kb = 0; kb < KDIM / 32; ++kb) {
                __builtin_prefetch(pB + (kb + 1) * 32, 0, 3);
                v16bf bf = load_fragB(pB, kb, hi);
#pragma unroll
                for (int mt = 0; mt < 4; ++mt) {
                    const unsigned short* pA = xh + (long)(mt * 16 + halfl) * KDIM;
                    v16bf af = load_fragA(pA, kb, hi);
                    acc[mt] = __builtin_amdgcn_wmma_f32_16x16x32_bf16(
                        false, af, false, bf, (short)0, acc[mt], false, false);
                }
            }
            // D layout: lane n=lane&15, rows m = (lane>=16?8:0)+i  (ISA 7.12.2)
#pragma unroll
            for (int mt = 0; mt < 4; ++mt)
#pragma unroll
                for (int i = 0; i < 8; ++i)
                    gbuf[mt * 16 + hi * 8 + i][nt * 16 + halfl] = acc[mt][i];
        }
        __syncthreads();

        // LSTM elementwise on owned slice (i,f,g,o gate order)
        for (int it = tid; it < BB * 32; it += NTHR) {
            int b = it >> 5, j = it & 31;
            int half = j >> 4, cc = j & 15;
            int gj = wg * 32 + j;
            float gi = gbuf[b][(0 * 2 + half) * 16 + cc] + bsum[0 * CDIM + gj];
            float gf = gbuf[b][(1 * 2 + half) * 16 + cc] + bsum[1 * CDIM + gj];
            float gg = gbuf[b][(2 * 2 + half) * 16 + cc] + bsum[2 * CDIM + gj];
            float go = gbuf[b][(3 * 2 + half) * 16 + cc] + bsum[3 * CDIM + gj];
            float ii = sigm(gi), ff = sigm(gf), g2 = tanhf(gg), oo = sigm(go);
            float cn = ff * cG[b * CDIM + gj] + ii * g2;
            float hn = oo * tanhf(cn);
            cG[b * CDIM + gj] = cn;
            hG[b * CDIM + gj] = hn;
            out[(long)t * BB * ODIM + (long)b * ODIM + gj] = hn;
        }
        grid_barrier(cnt, ++bar * NWG);

        // ========== Phase 2: NTM write/read, 4 batches per WG (float4 paths) ==========
        for (int bi = 0; bi < 4; ++bi) {
            int b = wg * 4 + bi;
            {   // co <- h[b] (b128 loads)
                const v4f* hp = (const v4f*)(hG + (long)b * CDIM);
                v4f* cp = (v4f*)co_s;
                for (int j = tid; j < CDIM / 4; j += NTHR) cp[j] = hp[j];
            }
            __syncthreads();

            // betas (vectorized partials + block reduce)
            float pw = 0.f, pr = 0.f;
            {
                const v4f* cp = (const v4f*)co_s;
                const v4f* bw = (const v4f*)w_bw;
                const v4f* br = (const v4f*)w_br;
                for (int j = tid; j < CDIM / 4; j += NTHR) {
                    v4f cv = cp[j];
                    pw += dot4(bw[j], cv);
                    pr += dot4(br[j], cv);
                }
            }
            float sw = block_sum(pw, red, tid);
            float sr = block_sum(pr, red, tid);
            if (tid == 0) { scal[0] = splus(sw); scal[1] = splus(sr); }

            // 4 mat-vecs, one 64-thread group each (float4 rows)
            {
                int grp = tid >> 6, j = tid & 63;
                const v4f* cp = (const v4f*)co_s;
                float a0 = 0.f;
                if (grp == 0) {
                    const v4f* wr = (const v4f*)(W_kw + (long)j * CDIM);
                    for (int k = 0; k < CDIM / 4; ++k) a0 += dot4(wr[k], cp[k]);
                    kv[j] = tanhf(a0 + b_kw[j]);
                } else if (grp == 1) {
                    const v4f* wr = (const v4f*)(W_e + (long)j * CDIM);
                    for (int k = 0; k < CDIM / 4; ++k) a0 += dot4(wr[k], cp[k]);
                    ev[j] = sigm(a0 + b_e[j]);
                } else if (grp == 2) {
                    const v4f* wr = (const v4f*)(W_kr + (long)j * CDIM);
                    for (int k = 0; k < CDIM / 4; ++k) a0 += dot4(wr[k], cp[k]);
                    krv[j] = tanhf(a0 + b_kr[j]);
                } else {
                    const v4f* wr = (const v4f*)(W_a + (long)j * IDIM);
                    const v4f* eb = (const v4f*)(embs + (long)t * BB * IDIM + (long)b * IDIM);
                    for (int k = 0; k < IDIM / 4; ++k) a0 += dot4(wr[k], eb[k]);
                    av[j] = tanhf(a0 + b_a[j]);
                }
            }
            __syncthreads();

            // write addressing (cosine + softmax)
            float kk = (tid < MM) ? kv[tid] * kv[tid] : 0.f;
            float kn = sqrtf(block_sum(kk, red, tid));
            float logit = -1e30f;
            if (tid < NN) {
                const v4f* mp = (const v4f*)(memG + ((long)b * NN + tid) * MM);
                const v4f* kp = (const v4f*)kv;
                float sim = 0.f, nn = 0.f;
                for (int m = 0; m < MM / 4; ++m) {
                    v4f v = mp[m];
                    sim += dot4(v, kp[m]);
                    nn  += dot4(v, v);
                }
                logit = scal[0] * sim / (sqrtf(nn) * kn + EPS);
            }
            softmax128(logit, wwv, red, tid);

            // erase / add (float4)
            for (int idx = tid; idx < NN * MM / 4; idx += NTHR) {
                int n = idx >> 4, m4 = idx & 15;
                v4f* p = (v4f*)(memG + ((long)b * NN + n) * MM) + m4;
                v4f v = *p;
                v4f e4 = ((const v4f*)ev)[m4];
                v4f a4 = ((const v4f*)av)[m4];
                float w = wwv[n];
                v4f o;
                o[0] = v[0] * (1.f - w * e4[0]) + w * a4[0];
                o[1] = v[1] * (1.f - w * e4[1]) + w * a4[1];
                o[2] = v[2] * (1.f - w * e4[2]) + w * a4[2];
                o[3] = v[3] * (1.f - w * e4[3]) + w * a4[3];
                *p = o;
            }
            __syncthreads();

            // read addressing on updated memory
            float rk = (tid < MM) ? krv[tid] * krv[tid] : 0.f;
            float rn = sqrtf(block_sum(rk, red, tid));
            float logit2 = -1e30f;
            if (tid < NN) {
                const v4f* mp = (const v4f*)(memG + ((long)b * NN + tid) * MM);
                const v4f* kp = (const v4f*)krv;
                float sim = 0.f, nn = 0.f;
                for (int m = 0; m < MM / 4; ++m) {
                    v4f v = mp[m];
                    sim += dot4(v, kp[m]);
                    nn  += dot4(v, v);
                }
                logit2 = scal[1] * sim / (sqrtf(nn) * rn + EPS);
            }
            softmax128(logit2, wwv, red, tid);    // wwv now holds wr

            // r = wr @ Mem; emit outputs + next-step bf16 activations
            if (tid < MM) {
                int m = tid;
                float rv = 0.f;
                const float* mb = memG + (long)b * NN * MM;
                for (int n = 0; n < NN; ++n) rv += wwv[n] * mb[n * MM + m];
                out[(long)t * BB * ODIM + (long)b * ODIM + CDIM + m] = rv;
                xh[(long)b * KDIM + IDIM + m] = f2bf(rv);
            }
            for (int j = tid; j < CDIM; j += NTHR)
                xh[(long)b * KDIM + IDIM + MM + j] = f2bf(co_s[j]);
            if (t + 1 < TT)
                for (int k = tid; k < IDIM; k += NTHR)
                    xh[(long)b * KDIM + k] = f2bf(embs[(long)(t + 1) * BB * IDIM + (long)b * IDIM + k]);
            __syncthreads();
        }
        grid_barrier(cnt, ++bar * NWG);
    }
}

// ---------------- launcher ----------------
extern "C" void kernel_launch(void* const* d_in, const int* in_sizes, int n_in,
                              void* d_out, int out_size, void* d_ws, size_t ws_size,
                              hipStream_t stream) {
    (void)in_sizes; (void)n_in; (void)out_size; (void)ws_size;
    const float* embs = (const float*)d_in[0];
    const float* W_ih = (const float*)d_in[1];
    const float* W_hh = (const float*)d_in[2];
    const float* b_ih = (const float*)d_in[3];
    const float* b_hh = (const float*)d_in[4];
    const float* W_kr = (const float*)d_in[5];
    const float* b_kr = (const float*)d_in[6];
    const float* w_br = (const float*)d_in[7];
    const float* W_kw = (const float*)d_in[8];
    const float* b_kw = (const float*)d_in[9];
    const float* w_bw = (const float*)d_in[10];
    const float* W_e  = (const float*)d_in[11];
    const float* b_e  = (const float*)d_in[12];
    const float* W_a  = (const float*)d_in[13];
    const float* b_a  = (const float*)d_in[14];
    const float* h0   = (const float*)d_in[15];
    const float* c0   = (const float*)d_in[16];
    const float* r0   = (const float*)d_in[17];
    const float* mem0 = (const float*)d_in[18];

    uint8_t* ws = (uint8_t*)d_ws;
    unsigned short* Wcat = (unsigned short*)(ws + OFF_WCAT);
    unsigned short* xh   = (unsigned short*)(ws + OFF_XH);
    float* hG   = (float*)(ws + OFF_H);
    float* cG   = (float*)(ws + OFF_C);
    float* memG = (float*)(ws + OFF_MEM);
    unsigned* cnt = (unsigned*)(ws + OFF_CNT);
    float* bsum = (float*)(ws + OFF_BS);
    float* out = (float*)d_out;

    mann_init_kernel<<<2048, 256, 0, stream>>>(embs, W_ih, W_hh, b_ih, b_hh,
                                               h0, c0, r0, mem0,
                                               Wcat, xh, hG, cG, memG, bsum, cnt);
    mann_step_kernel<<<NWG, NTHR, 0, stream>>>(embs, bsum, W_kr, b_kr, w_br,
                                               W_kw, b_kw, w_bw, W_e, b_e, W_a, b_a,
                                               Wcat, xh, hG, cG, memG, cnt, out);
}
